// GINConcat_67602785239437
// MI455X (gfx1250) — matrix operand order repaired and checked by reading, not compiled, not run on hardware
//
#include <hip/hip_runtime.h>
#include <hip/hip_bf16.h>

typedef __attribute__((ext_vector_type(16))) __bf16 v16bf;
typedef __attribute__((ext_vector_type(8)))  float  v8f;

#define N_NODES 50000
#define EDGES   800000
#define GRAPHS  512
#define F_IN    128
#define HID     256
#define LDIM    (3*HID + F_IN)   /* 896 */
#define BN_EPS  1e-5f

// ---------- helpers ----------
__device__ __forceinline__ unsigned short f2bf(float f) {
    unsigned u = __float_as_uint(f);
    unsigned r = (u + 0x7FFFu + ((u >> 16) & 1u)) >> 16;   // RNE
    return (unsigned short)r;
}

// ---------- tiny utility kernels ----------
__global__ void zero_f32(float* __restrict__ p, int n) {
    int i = blockIdx.x * blockDim.x + threadIdx.x;
    if (i < n) p[i] = 0.f;
}

__global__ void cvt_bf16(const float* __restrict__ s, unsigned short* __restrict__ d, int n) {
    int i = blockIdx.x * blockDim.x + threadIdx.x;
    if (i < n) d[i] = f2bf(s[i]);
}

// t = bf16(h + agg), two elements per thread (one dword store)
__global__ void prep_bf16(const float* __restrict__ h, const float* __restrict__ agg,
                          unsigned short* __restrict__ t, int n2) {
    int i = blockIdx.x * blockDim.x + threadIdx.x;
    if (i >= n2) return;
    int j = i * 2;
    unsigned lo = f2bf(h[j]     + agg[j]);
    unsigned hi = f2bf(h[j + 1] + agg[j + 1]);
    ((unsigned*)t)[i] = lo | (hi << 16);
}

// agg[dst[e]][c..c+3] += h[src[e]][c..c+3]
__global__ void scatter_add(const float* __restrict__ h, const int* __restrict__ src,
                            const int* __restrict__ dst, float* __restrict__ agg, int K4) {
    long long i = (long long)blockIdx.x * blockDim.x + threadIdx.x;
    long long tot = (long long)EDGES * K4;
    if (i >= tot) return;
    int e = (int)(i / K4);
    int c = (int)(i % K4) * 4;
    int K = K4 * 4;
    int s = src[e], d = dst[e];
    float4 v = *(const float4*)(h + (long long)s * K + c);
    float* a = agg + (long long)d * K + c;
    atomicAdd(a + 0, v.x); atomicAdd(a + 1, v.y);
    atomicAdd(a + 2, v.z); atomicAdd(a + 3, v.w);
}

// hcat[batch[node]][coloff + c..c+3] += h[node][c..c+3]
__global__ void pool_add(const float* __restrict__ h, const int* __restrict__ batch,
                         float* __restrict__ hcat, int K4, int coloff) {
    long long i = (long long)blockIdx.x * blockDim.x + threadIdx.x;
    long long tot = (long long)N_NODES * K4;
    if (i >= tot) return;
    int node = (int)(i / K4);
    int c = (int)(i % K4) * 4;
    int K = K4 * 4;
    int g = batch[node];
    float4 v = *(const float4*)(h + (long long)node * K + c);
    float* o = hcat + (long long)g * LDIM + coloff + c;
    atomicAdd(o + 0, v.x); atomicAdd(o + 1, v.y);
    atomicAdd(o + 2, v.z); atomicAdd(o + 3, v.w);
}

// ---------- WMMA GEMM:  out[M x Nout] = act( A[M x K] * W[Nout x K]^T + epilogue ) ----------
// MODE 0: bias + BN + ReLU  -> bf16 out
// MODE 1: bias + BN + ReLU  -> f32  out
// MODE 2: bias + ReLU       -> f32  out
// One wave computes a 32 x 64 output block: 2 M-tiles x 4 N-tiles.
// Each B fragment is reused by 2 WMMAs -> 12 b128 loads per 8 WMMAs.
// M-tile count may be odd: second tile index is clamped (uniform, race-free redundancy).
template <int MODE>
__global__ void wmma_gemm(const unsigned short* __restrict__ A,
                          const unsigned short* __restrict__ W,
                          float* __restrict__ outF, unsigned short* __restrict__ outB,
                          int M, int K, int Nout,
                          const float* __restrict__ bias,
                          const float* __restrict__ gam, const float* __restrict__ bet,
                          const float* __restrict__ rm,  const float* __restrict__ rv) {
    const int lane = threadIdx.x & 31;
    const int wave = threadIdx.x >> 5;
    const int tm  = M >> 4;              // 16-row tiles
    const int tm2 = (tm + 1) >> 1;       // pairs of M-tiles
    const int tn  = Nout >> 6;           // 64-col strips
    int unit = blockIdx.x * (blockDim.x >> 5) + wave;
    if (unit >= tm2 * tn) return;
    const int um2 = unit % tm2;
    const int un  = unit / tm2;

    const int ma = um2 * 2;
    const int mb = (ma + 1 < tm) ? (ma + 1) : (tm - 1);   // clamp (uniform)

    const int half = lane >> 4;
    const int l15  = lane & 15;
    const long long rowa = (long long)(ma * 16 + l15) * K;
    const long long rowb = (long long)(mb * 16 + l15) * K;
    const int ncol0 = un * 64 + l15;

    v8f accA0 = {0,0,0,0,0,0,0,0}, accA1 = accA0, accA2 = accA0, accA3 = accA0;
    v8f accB0 = accA0, accB1 = accA0, accB2 = accA0, accB3 = accA0;

    for (int k0 = 0; k0 < K; k0 += 32) {
        // A fragments: 16-bit A 16x32 layout (ISA 7.12.2)
        union { v16bf v; unsigned u[8]; } aA, aB;
        const unsigned short* apA = A + rowa + k0 + half * 8;
        const unsigned short* apB = A + rowb + k0 + half * 8;
        #pragma unroll
        for (int j = 0; j < 4; ++j) {
            aA.u[j]     = *(const unsigned*)(apA + 2 * j);
            aA.u[4 + j] = *(const unsigned*)(apA + 16 + 2 * j);
            aB.u[j]     = *(const unsigned*)(apB + 2 * j);
            aB.u[4 + j] = *(const unsigned*)(apB + 16 + 2 * j);
        }
        #pragma unroll
        for (int t = 0; t < 4; ++t) {
            // B fragment: lane owns column ncol0+16t; K pairs half*8 + 2j (j<4), 16 + half*8 + 2(j-4) (j>=4)
            union { v16bf v; unsigned u[8]; } b;
            const unsigned short* bp = W + (long long)(ncol0 + t * 16) * K + k0 + half * 8;
            #pragma unroll
            for (int j = 0; j < 8; ++j) b.u[j] = *(const unsigned*)(bp + 2 * j + (j >= 4 ? 8 : 0));
            v8f cA = (t == 0) ? accA0 : (t == 1) ? accA1 : (t == 2) ? accA2 : accA3;
            v8f cB = (t == 0) ? accB0 : (t == 1) ? accB1 : (t == 2) ? accB2 : accB3;
            cA = __builtin_amdgcn_wmma_f32_16x16x32_bf16(false, aA.v, false, b.v,
                                                         (short)0, cA, false, false);
            cB = __builtin_amdgcn_wmma_f32_16x16x32_bf16(false, aB.v, false, b.v,
                                                         (short)0, cB, false, false);
            if (t == 0) { accA0 = cA; accB0 = cB; }
            else if (t == 1) { accA1 = cA; accB1 = cB; }
            else if (t == 2) { accA2 = cA; accB2 = cB; }
            else             { accA3 = cA; accB3 = cB; }
        }
    }

    // epilogue: lane holds column n; rows tile*16 + half*8 + j
    #pragma unroll
    for (int t = 0; t < 4; ++t) {
        const v8f cA = (t == 0) ? accA0 : (t == 1) ? accA1 : (t == 2) ? accA2 : accA3;
        const v8f cB = (t == 0) ? accB0 : (t == 1) ? accB1 : (t == 2) ? accB2 : accB3;
        const int n = ncol0 + t * 16;
        float bs = bias[n];
        float scale = 1.f, shift = 0.f;
        if (MODE != 2) {
            float s = gam[n] * rsqrtf(rv[n] + BN_EPS);
            scale = s;
            shift = bet[n] - rm[n] * s;
        }
        #pragma unroll
        for (int j = 0; j < 8; ++j) {
            int mA = ma * 16 + half * 8 + j;
            int mB = mb * 16 + half * 8 + j;
            float vA = cA[j] + bs;
            float vB = cB[j] + bs;
            if (MODE != 2) { vA = vA * scale + shift; vB = vB * scale + shift; }
            vA = vA > 0.f ? vA : 0.f;
            vB = vB > 0.f ? vB : 0.f;
            long long oA = (long long)mA * Nout + n;
            long long oB = (long long)mB * Nout + n;
            if (MODE == 0) { outB[oA] = f2bf(vA); outB[oB] = f2bf(vB); }
            else           { outF[oA] = vA;       outF[oB] = vB; }
        }
    }
}

// out[g] = dot(z1[g,:], hw1) + hb1
__global__ void head_out(const float* __restrict__ z1, const float* __restrict__ hw1,
                         const float* __restrict__ hb1, float* __restrict__ out) {
    __shared__ float red[256];
    const float* row = z1 + (long long)blockIdx.x * LDIM;
    float s = 0.f;
    for (int c = threadIdx.x; c < LDIM; c += blockDim.x) s += row[c] * hw1[c];
    red[threadIdx.x] = s;
    __syncthreads();
    for (int off = 128; off > 0; off >>= 1) {
        if ((int)threadIdx.x < off) red[threadIdx.x] += red[threadIdx.x + off];
        __syncthreads();
    }
    if (threadIdx.x == 0) out[blockIdx.x] = red[0] + hb1[0];
}

// ---------- launch ----------
static inline int cdiv(long long a, long long b) { return (int)((a + b - 1) / b); }

extern "C" void kernel_launch(void* const* d_in, const int* in_sizes, int n_in,
                              void* d_out, int out_size, void* d_ws, size_t ws_size,
                              hipStream_t stream) {
    (void)in_sizes; (void)n_in; (void)out_size; (void)ws_size;
    const float* x     = (const float*)d_in[0];
    const int*   ei    = (const int*)d_in[1];
    const int*   batch = (const int*)d_in[2];
    const int*   esrc  = ei;
    const int*   edst  = ei + EDGES;
    const float* hw0 = (const float*)d_in[39];
    const float* hb0 = (const float*)d_in[40];
    const float* hw1 = (const float*)d_in[41];
    const float* hb1 = (const float*)d_in[42];

    char* ws = (char*)d_ws;
    size_t off = 0;
    float*          hbuf = (float*)(ws + off);          off += (size_t)N_NODES * HID * 4;
    float*          agg  = (float*)(ws + off);          off += (size_t)N_NODES * HID * 4;
    unsigned short* tA   = (unsigned short*)(ws + off); off += (size_t)N_NODES * HID * 2;
    unsigned short* wb0  = (unsigned short*)(ws + off); off += (size_t)HID * HID * 2;
    unsigned short* wb1  = (unsigned short*)(ws + off); off += (size_t)HID * HID * 2;
    float*          hcat = (float*)(ws + off);          off += (size_t)GRAPHS * LDIM * 4;
    unsigned short* hcb  = (unsigned short*)(ws + off); off += (size_t)GRAPHS * LDIM * 2;
    unsigned short* hw0b = (unsigned short*)(ws + off); off += (size_t)LDIM * LDIM * 2;
    float*          z1   = (float*)(ws + off);          off += (size_t)GRAPHS * LDIM * 4;
    unsigned short* m2   = (unsigned short*)agg;        // reuse agg after prep

    const int T = 256;

    // pools[0] = segment_sum(x)
    zero_f32<<<cdiv((long long)GRAPHS * LDIM, T), T, 0, stream>>>(hcat, GRAPHS * LDIM);
    pool_add<<<cdiv((long long)N_NODES * (F_IN / 4), T), T, 0, stream>>>(x, batch, hcat, F_IN / 4, 0);

    const float* hin = x;
    int Kin = F_IN;
    for (int i = 0; i < 3; ++i) {
        const int base = 3 + 12 * i;
        const float* w0  = (const float*)d_in[base + 0];
        const float* b0  = (const float*)d_in[base + 1];
        const float* bng = (const float*)d_in[base + 2];
        const float* bnb = (const float*)d_in[base + 3];
        const float* brm = (const float*)d_in[base + 4];
        const float* brv = (const float*)d_in[base + 5];
        const float* w1  = (const float*)d_in[base + 6];
        const float* b1  = (const float*)d_in[base + 7];
        const float* og  = (const float*)d_in[base + 8];
        const float* ob  = (const float*)d_in[base + 9];
        const float* orm = (const float*)d_in[base + 10];
        const float* orv = (const float*)d_in[base + 11];

        zero_f32<<<cdiv((long long)N_NODES * Kin, T), T, 0, stream>>>(agg, N_NODES * Kin);
        scatter_add<<<cdiv((long long)EDGES * (Kin / 4), T), T, 0, stream>>>(hin, esrc, edst, agg, Kin / 4);
        prep_bf16<<<cdiv((long long)N_NODES * Kin / 2, T), T, 0, stream>>>(hin, agg, tA, N_NODES * Kin / 2);

        cvt_bf16<<<cdiv((long long)HID * Kin, T), T, 0, stream>>>(w0, wb0, HID * Kin);
        {   // m2 = relu(bn(tA @ w0^T + b0))  -> bf16
            int tm2 = (N_NODES / 16 + 1) / 2;
            int units = tm2 * (HID / 64);
            wmma_gemm<0><<<cdiv(units, T / 32), T, 0, stream>>>(
                tA, wb0, nullptr, m2, N_NODES, Kin, HID, b0, bng, bnb, brm, brv);
        }
        cvt_bf16<<<cdiv((long long)HID * HID, T), T, 0, stream>>>(w1, wb1, HID * HID);
        {   // h = relu(bn(m2 @ w1^T + b1))  -> f32
            int tm2 = (N_NODES / 16 + 1) / 2;
            int units = tm2 * (HID / 64);
            wmma_gemm<1><<<cdiv(units, T / 32), T, 0, stream>>>(
                m2, wb1, hbuf, nullptr, N_NODES, HID, HID, b1, og, ob, orm, orv);
        }
        pool_add<<<cdiv((long long)N_NODES * (HID / 4), T), T, 0, stream>>>(
            hbuf, batch, hcat, HID / 4, F_IN + i * HID);

        hin = hbuf;
        Kin = HID;
    }

    // head: z1 = relu(hcat @ hw0^T + hb0)
    cvt_bf16<<<cdiv((long long)GRAPHS * LDIM, T), T, 0, stream>>>(hcat, hcb, GRAPHS * LDIM);
    cvt_bf16<<<cdiv((long long)LDIM * LDIM, T), T, 0, stream>>>(hw0, hw0b, LDIM * LDIM);
    {
        int tm2 = (GRAPHS / 16 + 1) / 2;
        int units = tm2 * (LDIM / 64);
        wmma_gemm<2><<<cdiv(units, T / 32), T, 0, stream>>>(
            hcb, hw0b, z1, nullptr, GRAPHS, LDIM, LDIM, hb0, nullptr, nullptr, nullptr, nullptr);
    }
    // out[g] = z1[g] . hw1 + hb1
    head_out<<<GRAPHS, 256, 0, stream>>>(z1, hw1, hb1, (float*)d_out);
}
